// SS2Dreg_48026324303964
// MI455X (gfx1250) — matrix operand order repaired and checked by reading, not compiled, not run on hardware
//
#include <hip/hip_runtime.h>
#include <hip/hip_bf16.h>
#include <cstdint>

typedef __attribute__((ext_vector_type(16))) _Float16 v16h;
typedef __attribute__((ext_vector_type(8)))  float    v8f;
typedef int v4i __attribute__((vector_size(16)));

#define DI    384      // D_INNER
#define NST   16       // D_STATE
#define RNK   12       // DT_RANK
#define LREGT 16       // LREG
#define L2D   4096     // H*W
#define LTOT  4112     // L2D + LREG
#define NCH   16       // scan chunks
#define CHL   257      // steps per chunk (16*257 = 4112)

#ifndef __has_builtin
#define __has_builtin(x) 0
#endif
#if __has_builtin(__builtin_amdgcn_global_load_async_to_lds_b128)
#define HAVE_ASYNC_LDS 1
#else
#define HAVE_ASYNC_LDS 0
#endif

__device__ __forceinline__ void cvt4(v16h& h, int o, float4 f) {
  h[o+0] = (_Float16)f.x; h[o+1] = (_Float16)f.y;
  h[o+2] = (_Float16)f.z; h[o+3] = (_Float16)f.w;
}

__device__ __forceinline__ int dir_pix(int k, int l) {
  if (k == 0) return l;
  if (k == 1) return ((l & 63) << 6) | (l >> 6);
  if (k == 2) return (L2D - 1) - l;
  int lm = (L2D - 1) - l;
  return ((lm & 63) << 6) | (lm >> 6);
}

// Stage n4 float4s from global to LDS, whole workgroup cooperates.
__device__ __forceinline__ void stage_to_lds(const float* g, float* l, int n4) {
  int tid = threadIdx.x, nt = blockDim.x;
#if HAVE_ASYNC_LDS
  __attribute__((address_space(1))) v4i* gp = (__attribute__((address_space(1))) v4i*)g;
  __attribute__((address_space(3))) v4i* lp = (__attribute__((address_space(3))) v4i*)l;
  for (int i = tid; i < n4; i += nt)
    __builtin_amdgcn_global_load_async_to_lds_b128(gp + i, lp + i, 0, 0);
#if __has_builtin(__builtin_amdgcn_s_wait_asynccnt)
  __builtin_amdgcn_s_wait_asynccnt(0);
#else
  asm volatile("s_wait_asynccnt 0" ::: "memory");
#endif
#else
  for (int i = tid; i < n4; i += nt)
    ((float4*)l)[i] = ((const float4*)g)[i];
#endif
  __syncthreads();
}

// ---------------------------------------------------------------------------
// Register-blocked WMMA GEMM: C[m,n] = sum_k A[m*lda+k]*B[n*ldb+k].
// One wave computes a 16x64 tile (4 accumulators share one A fragment).
// KDIM compile-time -> K loop fully unrolled (4 v_wmma per K step).
// ---------------------------------------------------------------------------
template <int KDIM>
__global__ void wmma_gemm_rr4(const float* __restrict__ A, int lda,
                              const float* __restrict__ Bm, int ldb,
                              float* __restrict__ C, int ldc, int nGroups) {
  int wid  = (blockIdx.x * blockDim.x + threadIdx.x) >> 5;
  int lane = threadIdx.x & 31;
  int mT = wid / nGroups, nG = wid % nGroups;
  int mn = lane & 15;
  bool hi = lane >= 16;
  const float* arow = A + (size_t)(mT * 16 + mn) * lda;
  const int klo  = hi ? 16 : 0;   // A frag: lane<16 -> K 0..15, lane>=16 -> K 16..31
  const int koff = hi ? 16 : 0;   // B frag: lane<16 -> K 0..15, lane>=16 -> K 16..31
  v8f acc0 = {}, acc1 = {}, acc2 = {}, acc3 = {};
#pragma unroll
  for (int kb = 0; kb < KDIM; kb += 32) {
    v16h av;
    cvt4(av, 0,  *(const float4*)(arow + kb + klo));
    cvt4(av, 4,  *(const float4*)(arow + kb + klo + 4));
    cvt4(av, 8,  *(const float4*)(arow + kb + klo + 8));
    cvt4(av, 12, *(const float4*)(arow + kb + klo + 12));
#pragma unroll
    for (int j = 0; j < 4; j++) {
      const float* brow = Bm + (size_t)(nG * 64 + j * 16 + mn) * ldb;
      v16h bv;
      cvt4(bv, 0,  *(const float4*)(brow + kb + koff));
      cvt4(bv, 4,  *(const float4*)(brow + kb + koff + 4));
      cvt4(bv, 8,  *(const float4*)(brow + kb + koff + 8));
      cvt4(bv, 12, *(const float4*)(brow + kb + koff + 12));
      v8f& acc = (j == 0) ? acc0 : (j == 1) ? acc1 : (j == 2) ? acc2 : acc3;
      acc = __builtin_amdgcn_wmma_f32_16x16x32_f16(false, av, false, bv,
                                                   (short)0, acc, false, false);
    }
  }
#pragma unroll
  for (int j = 0; j < 4; j++) {
    v8f& acc = (j == 0) ? acc0 : (j == 1) ? acc1 : (j == 2) ? acc2 : acc3;
#pragma unroll
    for (int r = 0; r < 8; r++) {
      int mr = (hi ? 8 : 0) + r;
      C[(size_t)(mT * 16 + mr) * ldc + nG * 64 + j * 16 + mn] = acc[r];
    }
  }
}

// ---------------------------------------------------------------------------
// x_dbl = einsum('bkdl,kcd->bkcl'); xs generated on the fly from xc/reg.
// One wave = 16(l) x 48(c) tile: strided A gather amortized across 3 WMMA tiles.
// Out: xdt[((b*4+k)*4112 + l)*48 + c] (c: 0..11 dts, 12..27 B, 28..43 C)
// ---------------------------------------------------------------------------
__global__ void xdbl_gemm(const float* __restrict__ xc, const float* __restrict__ reg,
                          const float* __restrict__ xpw, float* __restrict__ xdt) {
  int wid  = (blockIdx.x * blockDim.x + threadIdx.x) >> 5;  // 32*257 waves
  int lane = threadIdx.x & 31;
  int mT = wid % 257;
  int bk = wid / 257;
  int b = bk >> 2, k = bk & 3;
  int mn = lane & 15;
  bool hi = lane >= 16;
  const int klo  = hi ? 16 : 0;
  const int koff = hi ? 16 : 0;

  int l = mT * 16 + mn;
  const float* src;
  size_t stride;
  if (l < L2D) { src = xc + (((size_t)b * DI) << 12) + dir_pix(k, l); stride = 4096; }
  else         { src = reg + (l - L2D);                               stride = LREGT; }

  v8f acc0 = {}, acc1 = {}, acc2 = {};
#pragma unroll
  for (int kb = 0; kb < DI; kb += 32) {
    v16h av;
#pragma unroll
    for (int j = 0; j < 8; j++) {
      av[j]     = (_Float16)src[(size_t)(kb + klo + j) * stride];
      av[8 + j] = (_Float16)src[(size_t)(kb + klo + 8 + j) * stride];
    }
#pragma unroll
    for (int j = 0; j < 3; j++) {
      int c = j * 16 + mn;
      v16h bv;
      if (c < 44) {
        const float* brow = xpw + (size_t)(k * 44 + c) * DI;
        cvt4(bv, 0,  *(const float4*)(brow + kb + koff));
        cvt4(bv, 4,  *(const float4*)(brow + kb + koff + 4));
        cvt4(bv, 8,  *(const float4*)(brow + kb + koff + 8));
        cvt4(bv, 12, *(const float4*)(brow + kb + koff + 12));
      } else {
#pragma unroll
        for (int i = 0; i < 16; i++) bv[i] = (_Float16)0.f;
      }
      v8f& acc = (j == 0) ? acc0 : (j == 1) ? acc1 : acc2;
      acc = __builtin_amdgcn_wmma_f32_16x16x32_f16(false, av, false, bv,
                                                   (short)0, acc, false, false);
    }
  }
  float* out = xdt + (size_t)bk * LTOT * 48;
#pragma unroll
  for (int j = 0; j < 3; j++) {
    v8f& acc = (j == 0) ? acc0 : (j == 1) ? acc1 : acc2;
#pragma unroll
    for (int r = 0; r < 8; r++) {
      int lr = mT * 16 + (hi ? 8 : 0) + r;
      out[(size_t)lr * 48 + j * 16 + mn] = acc[r];
    }
  }
}

// ---------------------------------------------------------------------------
// LDS-tiled transpose: xv half of xz (B*L,768)[:,0:384] -> xcin[(b*384+d)*4096+l]
// ---------------------------------------------------------------------------
__global__ void transpose_xv(const float* __restrict__ xz, float* __restrict__ xcin) {
  __shared__ float tile[32][33];
  int col0 = blockIdx.x * 32;   // d
  int row0 = blockIdx.y * 32;   // g = b*4096 + l
#pragma unroll
  for (int i = 0; i < 4; i++) {
    int r = row0 + threadIdx.y + i * 8;
    tile[threadIdx.y + i * 8][threadIdx.x] = xz[(size_t)r * 768 + col0 + threadIdx.x];
  }
  __syncthreads();
#pragma unroll
  for (int i = 0; i < 4; i++) {
    int dd = col0 + threadIdx.y + i * 8;
    int g  = row0 + threadIdx.x;
    int b  = g >> 12, l = g & 4095;
    xcin[(((size_t)(b * DI + dd)) << 12) + l] = tile[threadIdx.x][threadIdx.y + i * 8];
  }
}

// ---------------------------------------------------------------------------
// Depthwise 3x3 conv (pad 1) + bias + SiLU, channel-major in/out.
// ---------------------------------------------------------------------------
__global__ void conv_dw_silu(const float* __restrict__ xcin, const float* __restrict__ cw,
                             const float* __restrict__ cb, float* __restrict__ xc) {
  int w = threadIdx.x, hh = blockIdx.x, d = blockIdx.y, b = blockIdx.z;
  const float* in = xcin + (((size_t)(b * DI + d)) << 12);
  const float* w9 = cw + d * 9;
  float acc = cb[d];
#pragma unroll
  for (int dy = -1; dy <= 1; dy++) {
    int y = hh + dy;
    if (y < 0 || y > 63) continue;
#pragma unroll
    for (int dx = -1; dx <= 1; dx++) {
      int x = w + dx;
      if (x < 0 || x > 63) continue;
      acc = fmaf(in[y * 64 + x], w9[(dy + 1) * 3 + (dx + 1)], acc);
    }
  }
  float sig = 1.f / (1.f + __expf(-acc));
  xc[(((size_t)(b * DI + d)) << 12) + hh * 64 + w] = acc * sig;
}

// ---------------------------------------------------------------------------
// Chunked selective scan, phase 1: per (b,k,d,chunk) run the recurrence with
// h0 = 0 and record (S = sum of delta, h_end[16]).  Since a_t = exp(delta*A_n),
// the chunk's state-decay product is exp(A_n * S) in closed form.
// One block = one (b,k,chunk): 384 threads, chunk rows staged in LDS (async).
// ---------------------------------------------------------------------------
__global__ void scan_pass1(const float* __restrict__ xdt, const float* __restrict__ xc,
                           const float* __restrict__ reg, const float* __restrict__ dtw,
                           const float* __restrict__ dtb, const float* __restrict__ Alog,
                           float* __restrict__ chsum) {
  __shared__ float rows[CHL * 48];
  int d  = threadIdx.x;
  int ch = blockIdx.x & (NCH - 1);
  int bk = blockIdx.x >> 4;
  int k = bk & 3, b = bk >> 2;
  stage_to_lds(xdt + ((size_t)bk * LTOT + ch * CHL) * 48, rows, CHL * 48 / 4);

  int row = k * DI + d;
  float An[NST];
#pragma unroll
  for (int n = 0; n < NST; n++) An[n] = -__expf(Alog[row * NST + n]);
  float bias = dtb[row];
  float wdt[RNK];
#pragma unroll
  for (int r = 0; r < RNK; r++) wdt[r] = dtw[row * RNK + r];
  float h[NST];
#pragma unroll
  for (int n = 0; n < NST; n++) h[n] = 0.f;
  float S = 0.f;

  const float* xcrow = xc + (((size_t)(b * DI + d)) << 12);
  int l0 = ch * CHL;
  for (int i = 0; i < CHL; i++) {
    const float* c48 = rows + i * 48;
    int l = l0 + i;
    float dta = bias;
#pragma unroll
    for (int r = 0; r < RNK; r++) dta = fmaf(c48[r], wdt[r], dta);
    float delta = fmaxf(dta, 0.f) + log1pf(__expf(-fabsf(dta)));
    S += delta;
    float u = (l < L2D) ? xcrow[dir_pix(k, l)] : reg[d * LREGT + (l - L2D)];
    float du = delta * u;
#pragma unroll
    for (int n = 0; n < NST; n++)
      h[n] = fmaf(h[n], __expf(delta * An[n]), du * c48[12 + n]);
  }
  float* o = chsum + ((size_t)(bk * DI + d) * NCH + ch) * 17;
  o[0] = S;
#pragma unroll
  for (int n = 0; n < NST; n++) o[1 + n] = h[n];
}

// ---------------------------------------------------------------------------
// Phase 2: sequential fixup over the 16 chunk boundaries (one thread per
// (b,k,d)): H_{c+1} = h_end_c + H_c * exp(A_n * S_c); store the H_c seeds.
// ---------------------------------------------------------------------------
__global__ void scan_fixup(const float* __restrict__ chsum, const float* __restrict__ Alog,
                           float* __restrict__ h0buf) {
  int t = blockIdx.x * blockDim.x + threadIdx.x;   // 12288 threads
  int d = t % DI, bk = t / DI;
  int k = bk & 3;
  int row = k * DI + d;
  float An[NST];
#pragma unroll
  for (int n = 0; n < NST; n++) An[n] = -__expf(Alog[row * NST + n]);
  float H[NST];
#pragma unroll
  for (int n = 0; n < NST; n++) H[n] = 0.f;
  const float* cs = chsum + (size_t)(bk * DI + d) * NCH * 17;
  float* hb = h0buf + (size_t)(bk * DI + d) * NCH * NST;
  for (int ch = 0; ch < NCH; ch++) {
#pragma unroll
    for (int n = 0; n < NST; n++) hb[ch * NST + n] = H[n];
    float S = cs[ch * 17];
#pragma unroll
    for (int n = 0; n < NST; n++)
      H[n] = fmaf(H[n], __expf(An[n] * S), cs[ch * 17 + 1 + n]);
  }
}

// ---------------------------------------------------------------------------
// Phase 3: replay each chunk from its corrected seed; emit outputs.
// Pixel outputs scatter into direction-corrected (b,pix,d,k) float4 slots
// (gather index == scatter index for each direction); reg tokens into racc.
// ---------------------------------------------------------------------------
__global__ void scan_pass2(const float* __restrict__ xdt, const float* __restrict__ xc,
                           const float* __restrict__ reg, const float* __restrict__ dtw,
                           const float* __restrict__ dtb, const float* __restrict__ Alog,
                           const float* __restrict__ Dsv, const float* __restrict__ h0buf,
                           float* __restrict__ ys4, float* __restrict__ racc) {
  __shared__ float rows[CHL * 48];
  int d  = threadIdx.x;
  int ch = blockIdx.x & (NCH - 1);
  int bk = blockIdx.x >> 4;
  int k = bk & 3, b = bk >> 2;
  stage_to_lds(xdt + ((size_t)bk * LTOT + ch * CHL) * 48, rows, CHL * 48 / 4);

  int row = k * DI + d;
  float An[NST];
#pragma unroll
  for (int n = 0; n < NST; n++) An[n] = -__expf(Alog[row * NST + n]);
  float Dv   = Dsv[row];
  float bias = dtb[row];
  float wdt[RNK];
#pragma unroll
  for (int r = 0; r < RNK; r++) wdt[r] = dtw[row * RNK + r];
  float h[NST];
  const float* hb = h0buf + (size_t)(bk * DI + d) * NCH * NST + ch * NST;
#pragma unroll
  for (int n = 0; n < NST; n++) h[n] = hb[n];

  const float* xcrow = xc + (((size_t)(b * DI + d)) << 12);
  int l0 = ch * CHL;
  for (int i = 0; i < CHL; i++) {
    const float* c48 = rows + i * 48;
    int l = l0 + i;
    float dta = bias;
#pragma unroll
    for (int r = 0; r < RNK; r++) dta = fmaf(c48[r], wdt[r], dta);
    float delta = fmaxf(dta, 0.f) + log1pf(__expf(-fabsf(dta)));
    float u;
    int pix = 0;
    if (l < L2D) { pix = dir_pix(k, l); u = xcrow[pix]; }
    else         { u = reg[d * LREGT + (l - L2D)]; }
    float du = delta * u;
    float y = 0.f;
#pragma unroll
    for (int n = 0; n < NST; n++) {
      h[n] = fmaf(h[n], __expf(delta * An[n]), du * c48[12 + n]);
      y = fmaf(h[n], c48[28 + n], y);
    }
    float o = y + Dv * u;
    if (l < L2D) ys4[(((size_t)(b << 12) + pix) * DI + d) * 4 + k] = o;
    else         racc[((size_t)(b * DI + d) * LREGT + (l - L2D)) * 4 + k] = o;
  }
}

// ---------------------------------------------------------------------------
// Merge 4 directions (contiguous float4) + LayerNorm over d + SiLU(z) gate.
// ---------------------------------------------------------------------------
__global__ void combine_ln_gate(const float* __restrict__ ys4, const float* __restrict__ xz,
                                const float* __restrict__ lnw, const float* __restrict__ lnb,
                                float* __restrict__ yg) {
  __shared__ float s1[DI], s2[DI];
  int bp = blockIdx.x;
  int d  = threadIdx.x;
  float4 v4 = *(const float4*)(ys4 + ((size_t)bp * DI + d) * 4);
  float v = (v4.x + v4.y) + (v4.z + v4.w);
  s1[d] = v; s2[d] = v * v;
  __syncthreads();
  for (int s = 192; s >= 3; s >>= 1) {
    if (d < s) { s1[d] += s1[d + s]; s2[d] += s2[d + s]; }
    __syncthreads();
  }
  float mean = (s1[0] + s1[1] + s1[2]) * (1.f / DI);
  float var  = (s2[0] + s2[1] + s2[2]) * (1.f / DI) - mean * mean;
  float rstd = rsqrtf(var + 1e-5f);
  float yn = (v - mean) * rstd * lnw[d] + lnb[d];
  float z  = xz[(size_t)bp * 768 + DI + d];
  float sig = 1.f / (1.f + __expf(-z));
  yg[(size_t)bp * DI + d] = yn * z * sig;
}

// ---------------------------------------------------------------------------
// Reg-token branch: r = sum_k, LayerNorm over d, project to 192 outs.
// ---------------------------------------------------------------------------
__global__ void r_ln_proj(const float* __restrict__ racc, const float* __restrict__ lnw,
                          const float* __restrict__ lnb, const float* __restrict__ wpr,
                          float* __restrict__ rout) {
  __shared__ float s1[DI], s2[DI], rn[DI];
  int b = blockIdx.x >> 4, lr = blockIdx.x & 15;
  int d = threadIdx.x;
  float4 v4 = *(const float4*)(racc + ((size_t)(b * DI + d) * LREGT + lr) * 4);
  float v = (v4.x + v4.y) + (v4.z + v4.w);
  s1[d] = v; s2[d] = v * v;
  __syncthreads();
  for (int s = 192; s >= 3; s >>= 1) {
    if (d < s) { s1[d] += s1[d + s]; s2[d] += s2[d + s]; }
    __syncthreads();
  }
  float mean = (s1[0] + s1[1] + s1[2]) * (1.f / DI);
  float var  = (s2[0] + s2[1] + s2[2]) * (1.f / DI) - mean * mean;
  float rstd = rsqrtf(var + 1e-5f);
  rn[d] = (v - mean) * rstd * lnw[d] + lnb[d];
  __syncthreads();
  if (d < 192) {
    float acc = 0.f;
    const float* wr = wpr + (size_t)d * DI;
    for (int dd = 0; dd < DI; dd++) acc = fmaf(rn[dd], wr[dd], acc);
    rout[(size_t)(b * 192 + d) * LREGT + lr] = acc;
  }
}

extern "C" void kernel_launch(void* const* d_in, const int* in_sizes, int n_in,
                              void* d_out, int out_size, void* d_ws, size_t ws_size,
                              hipStream_t stream) {
  (void)in_sizes; (void)n_in; (void)out_size; (void)ws_size;
  const float* x    = (const float*)d_in[0];
  const float* ipw  = (const float*)d_in[1];
  const float* cw   = (const float*)d_in[2];
  const float* cb   = (const float*)d_in[3];
  const float* xpw  = (const float*)d_in[4];
  const float* dtw  = (const float*)d_in[5];
  const float* dtb  = (const float*)d_in[6];
  const float* Alog = (const float*)d_in[7];
  const float* Dsv  = (const float*)d_in[8];
  const float* lnw  = (const float*)d_in[9];
  const float* lnb  = (const float*)d_in[10];
  const float* lnrw = (const float*)d_in[11];
  const float* lnrb = (const float*)d_in[12];
  const float* opw  = (const float*)d_in[13];
  const float* oprw = (const float*)d_in[14];
  const float* reg  = (const float*)d_in[15];

  // Workspace layout (floats), ~455 MB total; xcin reused as yg after conv.
  float* ws    = (float*)d_ws;
  float* xz    = ws;                        // 32768*768        = 25,165,824
  float* xcin  = xz    + 25165824;          // 8*384*4096       = 12,582,912
  float* xc    = xcin  + 12582912;          // 8*384*4096       = 12,582,912
  float* xdt   = xc    + 12582912;          // 32*4112*48       =  6,316,032
  float* ys4   = xdt   + 6316032;           // 8*4096*384*4     = 50,331,648
  float* racc  = ys4   + 50331648;          // 8*384*16*4       =    196,608
  float* chsum = racc  + 196608;            // 32*384*16*17     =  3,342,336
  float* h0buf = chsum + 3342336;           // 32*384*16*16     =  3,145,728
  float* yg    = xcin;                      // alias

  float* outp  = (float*)d_out;             // (8,64,64,192)
  float* routp = outp + 8 * 64 * 64 * 192;  // (8,192,16)

  // 1) xz = x @ in_proj_w^T : M=32768 (2048 mT), N=768 (12 nG of 64), K=192
  wmma_gemm_rr4<192><<<3072, 256, 0, stream>>>(x, 192, ipw, 192, xz, 768, 12);
  // 2) transpose xv half to channel-major
  transpose_xv<<<dim3(12, 1024), dim3(32, 8), 0, stream>>>(xz, xcin);
  // 3) depthwise conv + SiLU
  conv_dw_silu<<<dim3(64, DI, 8), 64, 0, stream>>>(xcin, cw, cb, xc);
  // 4) x_dbl GEMM: 32 (b,k) * 257 mT waves, 16x48 per wave
  xdbl_gemm<<<1028, 256, 0, stream>>>(xc, reg, xpw, xdt);
  // 5) chunked scan: partials -> boundary fixup -> replay with seeds
  scan_pass1<<<512, DI, 0, stream>>>(xdt, xc, reg, dtw, dtb, Alog, chsum);
  scan_fixup<<<96, 128, 0, stream>>>(chsum, Alog, h0buf);
  scan_pass2<<<512, DI, 0, stream>>>(xdt, xc, reg, dtw, dtb, Alog, Dsv, h0buf, ys4, racc);
  // 6) merge + LN + gate
  combine_ln_gate<<<32768, DI, 0, stream>>>(ys4, xz, lnw, lnb, yg);
  // 7) out = yg @ out_proj_w^T : M=32768, N=192 (3 nG of 64), K=384
  wmma_gemm_rr4<384><<<768, 256, 0, stream>>>(yg, DI, opw, DI, outp, 192, 3);
  // 8) reg branch
  r_ln_proj<<<128, DI, 0, stream>>>(racc, lnrw, lnrb, oprw, routp);
}
